// PointNetSetUpConv_42073499631910
// MI455X (gfx1250) — compile-verified, matrix operand order checked by reading, not run on hardware
//
#include <hip/hip_runtime.h>
#include <hip/hip_bf16.h>

typedef __attribute__((ext_vector_type(16))) _Float16 v16h;
typedef __attribute__((ext_vector_type(8)))  _Float16 v8h;
typedef __attribute__((ext_vector_type(8)))  float    v8f;
typedef __attribute__((ext_vector_type(4)))  unsigned int u32x4;
typedef __attribute__((ext_vector_type(8)))  int i32x8;
typedef __attribute__((ext_vector_type(4)))  int i32x4;

#define B_    4
#define N1_   8192
#define N2_   2048
#define C1_   256
#define C2_   256
#define KNN_  16
#define COUT_ 256
#define BN_EPS 1e-5f

#if defined(__has_builtin)
#if __has_builtin(__builtin_amdgcn_tensor_load_to_lds) && \
    __has_builtin(__builtin_amdgcn_s_wait_tensorcnt)
#define HAVE_TDM 1
#endif
#endif
#ifndef HAVE_TDM
#define HAVE_TDM 0
#endif

// ---------------------------------------------------------------------------
// KNN: one thread per query point; pos2 staged in LDS; register top-16.
// ---------------------------------------------------------------------------
__global__ void __launch_bounds__(256) knn_kernel(const float* __restrict__ pos1,
                                                  const float* __restrict__ pos2,
                                                  int* __restrict__ idx) {
  __shared__ float sx[N2_], sy[N2_], sz[N2_];
  const int b = blockIdx.x >> 5;
  const int n = ((blockIdx.x & 31) << 8) + threadIdx.x;
  for (int j = threadIdx.x; j < N2_; j += 256) {
    sx[j] = pos2[(b * 3 + 0) * N2_ + j];
    sy[j] = pos2[(b * 3 + 1) * N2_ + j];
    sz[j] = pos2[(b * 3 + 2) * N2_ + j];
  }
  __syncthreads();
  const float px = pos1[(b * 3 + 0) * N1_ + n];
  const float py = pos1[(b * 3 + 1) * N1_ + n];
  const float pz = pos1[(b * 3 + 2) * N1_ + n];
  float bd[KNN_]; int bi[KNN_];
#pragma unroll
  for (int k = 0; k < KNN_; ++k) { bd[k] = 3.4e38f; bi[k] = 0; }
  for (int j = 0; j < N2_; ++j) {
    float dx = sx[j] - px, dy = sy[j] - py, dz = sz[j] - pz;
    float d = dx * dx + dy * dy + dz * dz;
    if (d < bd[KNN_ - 1]) {
      float cd = d; int ci = j;
#pragma unroll
      for (int k = 0; k < KNN_; ++k) {
        if (cd < bd[k]) {
          float td = bd[k]; int ti = bi[k];
          bd[k] = cd; bi[k] = ci; cd = td; ci = ti;
        }
      }
    }
  }
  const size_t base = ((size_t)b * N1_ + n) * KNN_;
#pragma unroll
  for (int k = 0; k < KNN_; ++k) idx[base + k] = bi[k];
}

// ---------------------------------------------------------------------------
// Pre-convert weights to zero-padded f16: W[256 x Cin] f32 -> Wh[256 x KT32]
// ---------------------------------------------------------------------------
__global__ void __launch_bounds__(256) wcvt_kernel(const float* __restrict__ W,
                                                   int Cin, int KT32,
                                                   _Float16* __restrict__ Wh) {
  const int e = blockIdx.x * 256 + threadIdx.x;   // 256*KT32 total
  const int o = e / KT32, c = e % KT32;
  Wh[e] = (c < Cin) ? (_Float16)W[(size_t)o * Cin + c] : (_Float16)0.0f;
}

// ---------------------------------------------------------------------------
// TDM: async 2-D f16 tile load Global->LDS with row padding to ASTRIDE.
// Tile rows are 64B; pad_interval=3 (16 DWORDs) + pad_amount=3 (4 DWORDs)
// -> LDS row stride 80B == 40 halfs (conflict-free for fragment loads).
// clang-23 toolchain: 6-arg builtin (u32x4, i32x8, i32x4, i32x4, i32x8, cpol).
// ---------------------------------------------------------------------------
#if HAVE_TDM
__device__ __forceinline__ void tdm_load_2d_f16(
    unsigned lds_byte, const void* gptr, unsigned tensor_d0, unsigned tensor_d1,
    unsigned tile_d0, unsigned tile_d1, unsigned row_stride_elems) {
  const unsigned long long ga = (unsigned long long)(uintptr_t)gptr;
  u32x4 g0;
  g0[0] = 1u;                                             // count=1, no gather
  g0[1] = lds_byte;                                       // lds_addr (bytes)
  g0[2] = (unsigned)ga;                                   // global_addr[31:0]
  g0[3] = (unsigned)((ga >> 32) & 0x01FFFFFFu) | (2u << 30);  // addr[56:32], type=2
  i32x8 g1;
  g1[0] = (int)((1u << 16) |                              // data_size = 2B
                (1u << 20) |                              // pad_enable
                (3u << 22) |                              // pad_interval: 16 DW
                (3u << 25));                              // pad_amount: 4 DW
  g1[1] = (int)((tensor_d0 & 0xFFFFu) << 16);
  g1[2] = (int)(((tensor_d0 >> 16) & 0xFFFFu) | ((tensor_d1 & 0xFFFFu) << 16));
  g1[3] = (int)(((tensor_d1 >> 16) & 0xFFFFu) | ((tile_d0 & 0xFFFFu) << 16));
  g1[4] = (int)(tile_d1 & 0xFFFFu);                       // tile_dim1 | tile_dim2=0
  g1[5] = (int)row_stride_elems;                          // tensor_dim0_stride lo
  g1[6] = 0;                                              // stride hi | dim1_stride lo
  g1[7] = 0;
  i32x4 z4; z4[0] = 0; z4[1] = 0; z4[2] = 0; z4[3] = 0;
  i32x8 z8;
#pragma unroll
  for (int i = 0; i < 8; ++i) z8[i] = 0;
  __builtin_amdgcn_tensor_load_to_lds(g0, g1, z4, z4, z8, 0);
}
#endif

// ---------------------------------------------------------------------------
// WMMA fragment load from LDS row-major f16 tile (stride 40 halfs).
// ---------------------------------------------------------------------------
#define ASTRIDE 40
__device__ __forceinline__ v16h load_frag(const _Float16* base, int lane) {
  const int row = lane & 15;
  const int kb  = (lane >> 4) << 3;              // 0 or 8
  const _Float16* p = base + row * ASTRIDE;
  v8h lo = *(const v8h*)(p + kb);
  v8h hi = *(const v8h*)(p + 16 + kb);
  v16h r;
#pragma unroll
  for (int i = 0; i < 8; ++i) { r[i] = lo[i]; r[8 + i] = hi[i]; }
  return r;
}

// ---------------------------------------------------------------------------
// Fused GEMM: D[M x 256] = A[M x Cin] * W[256 x Cin]^T   (WMMA f16 -> f32)
//   MODE 0: A gathered on the fly: direction(3) ++ feature2 gather (GEMM1)
//   MODE 1: A = relu(Aprev * bnScale + bnShift), Aprev f16   (GEMM2 / GEMM4)
//   MODE 2: A = Aprev f16 as-is (TDM DMA when available)      (GEMM3)
// Weights come pre-converted/padded (Wh f16, width KT*32); the B tile is
// DMA'd by the Tensor Data Mover when available.
// ---------------------------------------------------------------------------
template <int MODE, int KT>
__global__ void __launch_bounds__(256) gemm_kernel(
    const _Float16* __restrict__ Ain,
    const float* __restrict__ bnScale, const float* __restrict__ bnShift,
    const int* __restrict__ knn, const float* __restrict__ pos1,
    const float* __restrict__ pos2, const float* __restrict__ feat2,
    const _Float16* __restrict__ Wh, const int Cin,
    _Float16* __restrict__ Yout,
    float* __restrict__ psum, float* __restrict__ psumsq) {
  constexpr int KT32 = KT * 32;
  __shared__ __align__(16) _Float16 sA[64 * ASTRIDE];
  __shared__ __align__(16) _Float16 sB[256 * ASTRIDE];
  __shared__ float sPart[4 * 256];
  __shared__ float sPartSq[4 * 256];
  __shared__ float sScale[256], sShift[256];
  __shared__ int   sIdx[64];
  __shared__ float sP1[3 * 64];

  const int tid  = threadIdx.x;
  const int wave = tid >> 5, lane = tid & 31;
  const int mt = wave >> 1;                      // 0..3 : 16-row M tile
  const int nh = wave & 1;                       // 0..1 : 128-col N half
  const int bb = (int)(((long)blockIdx.x * 64) >> 17);  // batch (MODE 0)

  if (MODE == 0) {
    if (tid < 64) {
      const int gm = blockIdx.x * 64 + tid;
      sIdx[tid] = knn[gm];
      const int n = (gm >> 4) & (N1_ - 1);
      sP1[0 * 64 + tid] = pos1[(bb * 3 + 0) * N1_ + n];
      sP1[1 * 64 + tid] = pos1[(bb * 3 + 1) * N1_ + n];
      sP1[2 * 64 + tid] = pos1[(bb * 3 + 2) * N1_ + n];
    }
  }
  if (MODE == 1) { sScale[tid] = bnScale[tid]; sShift[tid] = bnShift[tid]; }

  v8f acc[8];
#pragma unroll
  for (int i = 0; i < 8; ++i) {
#pragma unroll
    for (int j = 0; j < 8; ++j) acc[i][j] = 0.0f;
  }

  for (int ks = 0; ks < KT; ++ks) {
    const int cb = ks * 32;
    __syncthreads();

#if HAVE_TDM
    if (wave == 0) {
      // DMA weight tile [256 x 32] (row 64B) into padded sB
      tdm_load_2d_f16((unsigned)(uintptr_t)(void*)sB,
                      (const char*)Wh + (size_t)cb * 2,
                      (unsigned)KT32, 256u, 32u, 256u, (unsigned)KT32);
      if (MODE == 2) {
        // DMA activation tile [64 x 32] into padded sA
        tdm_load_2d_f16((unsigned)(uintptr_t)(void*)sA,
                        (const char*)(Ain + (size_t)blockIdx.x * 64 * Cin + cb),
                        (unsigned)Cin, (1u << 20), 32u, 64u, (unsigned)Cin);
      }
    }
#endif

    // ---- stage A tile (64 x 32) ----
    if (MODE == 0) {
#pragma unroll
      for (int i = 0; i < 8; ++i) {
        const int e = i * 256 + tid;
        const int r = e >> 5, c = e & 31;
        const int cg = cb + c;
        float v = 0.0f;
        if (cg < 3) {
          v = pos2[(bb * 3 + cg) * N2_ + sIdx[r]] - sP1[cg * 64 + r];
        } else if (cg < Cin) {
          v = feat2[((size_t)bb * C2_ + (cg - 3)) * N2_ + sIdx[r]];
        }
        sA[r * ASTRIDE + c] = (_Float16)v;
      }
    } else if (MODE == 1) {
      const int r = tid >> 2, ck = (tid & 3) << 3;   // one b128 per thread
      const v8h a = *(const v8h*)(Ain + (size_t)(blockIdx.x * 64 + r) * Cin + cb + ck);
      v8h h;
#pragma unroll
      for (int ii = 0; ii < 8; ++ii) {
        const int cg = cb + ck + ii;
        h[ii] = (_Float16)fmaxf((float)a[ii] * sScale[cg] + sShift[cg], 0.0f);
      }
      *(v8h*)(sA + r * ASTRIDE + ck) = h;
    } else {
#if !HAVE_TDM
      const int r = tid >> 2, ck = (tid & 3) << 3;
      *(v8h*)(sA + r * ASTRIDE + ck) =
          *(const v8h*)(Ain + (size_t)(blockIdx.x * 64 + r) * Cin + cb + ck);
#endif
    }

#if !HAVE_TDM
    // ---- stage W tile (256 x 32) from padded f16 weights: 4 b128 / thread ----
#pragma unroll
    for (int i = 0; i < 4; ++i) {
      const int e = i * 256 + tid;
      const int o = e >> 2, ck = (e & 3) << 3;
      *(v8h*)(sB + o * ASTRIDE + ck) = *(const v8h*)(Wh + (size_t)o * KT32 + cb + ck);
    }
#else
    if (wave == 0) __builtin_amdgcn_s_wait_tensorcnt((short)0);
#endif
    __syncthreads();

    // ---- 8 WMMAs per wave per k-step ----
    const v16h af = load_frag(&sA[(mt * 16) * ASTRIDE], lane);
#pragma unroll
    for (int nt = 0; nt < 8; ++nt) {
      const v16h bf = load_frag(&sB[(nh * 128 + nt * 16) * ASTRIDE], lane);
      acc[nt] = __builtin_amdgcn_wmma_f32_16x16x32_f16(
          false, af, false, bf, (short)0, acc[nt], false, false);
    }
  }

  // ---- deterministic per-channel partial stats ----
#pragma unroll
  for (int nt = 0; nt < 8; ++nt) {
    float s = 0.0f, sq = 0.0f;
#pragma unroll
    for (int r = 0; r < 8; ++r) { const float v = acc[nt][r]; s += v; sq += v * v; }
    s  += __shfl_xor(s, 16);
    sq += __shfl_xor(sq, 16);
    if (lane < 16) {
      const int ch = nh * 128 + nt * 16 + lane;
      sPart[mt * 256 + ch]   = s;
      sPartSq[mt * 256 + ch] = sq;
    }
  }
  __syncthreads();
  {
    const int ch = tid;
    const float s  = sPart[ch] + sPart[256 + ch] + sPart[512 + ch] + sPart[768 + ch];
    const float sq = sPartSq[ch] + sPartSq[256 + ch] + sPartSq[512 + ch] + sPartSq[768 + ch];
    psum[(size_t)blockIdx.x * 256 + ch]   = s;
    psumsq[(size_t)blockIdx.x * 256 + ch] = sq;
  }

  // ---- store pre-BN activations as f16 ----
  const int lhi = lane >> 4;
  const int ln  = lane & 15;
#pragma unroll
  for (int nt = 0; nt < 8; ++nt) {
    const int ch = nh * 128 + nt * 16 + ln;
#pragma unroll
    for (int r = 0; r < 8; ++r) {
      const int gm = blockIdx.x * 64 + mt * 16 + lhi * 8 + r;
      Yout[(size_t)gm * COUT_ + ch] = (_Float16)acc[nt][r];
    }
  }
}

// ---------------------------------------------------------------------------
// Fold BN into per-channel scale/shift (deterministic fixed-order reduce).
// ---------------------------------------------------------------------------
__global__ void __launch_bounds__(256) bn_finalize(
    const float* __restrict__ psum, const float* __restrict__ psumsq, int nblocks,
    const float* __restrict__ gamma, const float* __restrict__ beta,
    float invCount, float* __restrict__ scale, float* __restrict__ shift) {
  const int c = threadIdx.x;
  float s = 0.0f, sq = 0.0f;
  for (int b = 0; b < nblocks; ++b) {
    s  += psum[(size_t)b * 256 + c];
    sq += psumsq[(size_t)b * 256 + c];
  }
  const float mean = s * invCount;
  const float var  = sq * invCount - mean * mean;
  const float sc   = gamma[c] * rsqrtf(var + BN_EPS);
  scale[c] = sc;
  shift[c] = beta[c] - mean * sc;
}

// ---------------------------------------------------------------------------
// BN2+ReLU on Y2, max over 16 neighbors, concat feature1 -> A3 [32768 x 512]
// ---------------------------------------------------------------------------
__global__ void __launch_bounds__(256) max_concat_kernel(
    const _Float16* __restrict__ Y2, const float* __restrict__ s2,
    const float* __restrict__ t2, const float* __restrict__ feat1,
    _Float16* __restrict__ A3) {
  const int gid = blockIdx.x * 256 + threadIdx.x;
  const int ch  = gid & 511;
  const int row = gid >> 9;
  if (ch < 256) {
    const float sc = s2[ch], sh = t2[ch];
    float m = 0.0f;  // max_k relu(x_k) == max(0, max_k x_k)
    const size_t base = (size_t)row * KNN_ * COUT_ + ch;
#pragma unroll
    for (int k = 0; k < KNN_; ++k)
      m = fmaxf(m, (float)Y2[base + (size_t)k * COUT_] * sc + sh);
    A3[(size_t)row * 512 + ch] = (_Float16)m;
  } else {
    const int c = ch - 256;
    const int b = row >> 13, n = row & (N1_ - 1);
    A3[(size_t)row * 512 + ch] = (_Float16)feat1[((size_t)b * C1_ + c) * N1_ + n];
  }
}

// ---------------------------------------------------------------------------
// BN4 + ReLU on Y4, transpose-store to output [B, 256, N1] f32
// ---------------------------------------------------------------------------
__global__ void __launch_bounds__(256) output_kernel(
    const _Float16* __restrict__ Y4, const float* __restrict__ s4,
    const float* __restrict__ t4, float* __restrict__ out) {
  const long gid = (long)blockIdx.x * 256 + threadIdx.x;
  const int n  = (int)(gid & (N1_ - 1));
  const int t  = (int)(gid >> 13);
  const int ch = t & 255;
  const int b  = t >> 8;
  const float v = fmaxf(
      (float)Y4[((size_t)(b * N1_ + n)) * COUT_ + ch] * s4[ch] + t4[ch], 0.0f);
  out[((size_t)b * COUT_ + ch) * N1_ + n] = v;
}

// ---------------------------------------------------------------------------
// Launch pipeline
// ---------------------------------------------------------------------------
extern "C" void kernel_launch(void* const* d_in, const int* in_sizes, int n_in,
                              void* d_out, int out_size, void* d_ws, size_t ws_size,
                              hipStream_t stream) {
  const float* pos1  = (const float*)d_in[0];
  const float* pos2  = (const float*)d_in[1];
  const float* feat1 = (const float*)d_in[2];
  const float* feat2 = (const float*)d_in[3];
  const float* w11 = (const float*)d_in[4];   // [256, 259]
  const float* g11 = (const float*)d_in[5];
  const float* b11 = (const float*)d_in[6];
  const float* w12 = (const float*)d_in[7];   // [256, 256]
  const float* g12 = (const float*)d_in[8];
  const float* b12 = (const float*)d_in[9];
  const float* w21 = (const float*)d_in[10];  // [256, 512]
  const float* g21 = (const float*)d_in[11];
  const float* b21 = (const float*)d_in[12];
  const float* w22 = (const float*)d_in[13];  // [256, 256]
  const float* g22 = (const float*)d_in[14];
  const float* b22 = (const float*)d_in[15];
  float* out = (float*)d_out;

  const size_t M1 = (size_t)B_ * N1_ * KNN_;   // 524288
  const size_t M2 = (size_t)B_ * N1_;          // 32768
  const int NB1 = (int)(M1 / 64);              // 8192 blocks
  const int NB2 = (int)(M2 / 64);              // 512 blocks

  char* ws = (char*)d_ws;
  size_t o = 0;
  auto carve = [&](size_t bytes) {
    char* p = ws + o;
    o = (o + bytes + 255) & ~(size_t)255;
    return p;
  };
  int*      knn = (int*)      carve(M1 * sizeof(int));
  _Float16* Y1  = (_Float16*) carve(M1 * COUT_ * 2);
  _Float16* Y2  = (_Float16*) carve(M1 * COUT_ * 2);
  _Float16* A3  = (_Float16*) carve(M2 * 512 * 2);
  _Float16* Y3  = (_Float16*) carve(M2 * COUT_ * 2);
  _Float16* Y4  = (_Float16*) carve(M2 * COUT_ * 2);
  _Float16* Wh1 = (_Float16*) carve((size_t)256 * 288 * 2);
  _Float16* Wh2 = (_Float16*) carve((size_t)256 * 256 * 2);
  _Float16* Wh3 = (_Float16*) carve((size_t)256 * 512 * 2);
  _Float16* Wh4 = (_Float16*) carve((size_t)256 * 256 * 2);
  float* psum1   = (float*) carve((size_t)NB1 * 256 * 4);
  float* psumsq1 = (float*) carve((size_t)NB1 * 256 * 4);
  float* psum2   = (float*) carve((size_t)NB1 * 256 * 4);
  float* psumsq2 = (float*) carve((size_t)NB1 * 256 * 4);
  float* psum3   = (float*) carve((size_t)NB2 * 256 * 4);
  float* psumsq3 = (float*) carve((size_t)NB2 * 256 * 4);
  float* psum4   = (float*) carve((size_t)NB2 * 256 * 4);
  float* psumsq4 = (float*) carve((size_t)NB2 * 256 * 4);
  float* scale1 = (float*) carve(256 * 4); float* shift1 = (float*) carve(256 * 4);
  float* scale2 = (float*) carve(256 * 4); float* shift2 = (float*) carve(256 * 4);
  float* scale3 = (float*) carve(256 * 4); float* shift3 = (float*) carve(256 * 4);
  float* scale4 = (float*) carve(256 * 4); float* shift4 = (float*) carve(256 * 4);

  // 0) pre-convert weights to padded f16 (+KNN, independent)
  wcvt_kernel<<<288, 256, 0, stream>>>(w11, 259, 288, Wh1);
  wcvt_kernel<<<256, 256, 0, stream>>>(w12, 256, 256, Wh2);
  wcvt_kernel<<<512, 256, 0, stream>>>(w21, 512, 512, Wh3);
  wcvt_kernel<<<256, 256, 0, stream>>>(w22, 256, 256, Wh4);
  knn_kernel<<<(B_ * N1_) / 256, 256, 0, stream>>>(pos1, pos2, knn);

  // 1) GEMM1: gather(direction ++ feature2) @ W11^T  (Cin=259, KT=9)
  gemm_kernel<0, 9><<<NB1, 256, 0, stream>>>(
      nullptr, nullptr, nullptr, knn, pos1, pos2, feat2, Wh1, 259, Y1, psum1, psumsq1);
  bn_finalize<<<1, 256, 0, stream>>>(psum1, psumsq1, NB1, g11, b11,
                                     1.0f / (float)M1, scale1, shift1);

  // 2) GEMM2: relu(bn(Y1)) @ W12^T  (Cin=256, KT=8)
  gemm_kernel<1, 8><<<NB1, 256, 0, stream>>>(
      Y1, scale1, shift1, nullptr, nullptr, nullptr, nullptr, Wh2, 256, Y2, psum2, psumsq2);
  bn_finalize<<<1, 256, 0, stream>>>(psum2, psumsq2, NB1, g12, b12,
                                     1.0f / (float)M1, scale2, shift2);

  // 3) BN2+ReLU, max over K, concat feature1
  max_concat_kernel<<<(int)(M2 * 512 / 256), 256, 0, stream>>>(Y2, scale2, shift2, feat1, A3);

  // 4) GEMM3: A3 @ W21^T  (Cin=512, KT=16)
  gemm_kernel<2, 16><<<NB2, 256, 0, stream>>>(
      A3, nullptr, nullptr, nullptr, nullptr, nullptr, nullptr, Wh3, 512, Y3, psum3, psumsq3);
  bn_finalize<<<1, 256, 0, stream>>>(psum3, psumsq3, NB2, g21, b21,
                                     1.0f / (float)M2, scale3, shift3);

  // 5) GEMM4: relu(bn(Y3)) @ W22^T  (Cin=256, KT=8)
  gemm_kernel<1, 8><<<NB2, 256, 0, stream>>>(
      Y3, scale3, shift3, nullptr, nullptr, nullptr, nullptr, Wh4, 256, Y4, psum4, psumsq4);
  bn_finalize<<<1, 256, 0, stream>>>(psum4, psumsq4, NB2, g22, b22,
                                     1.0f / (float)M2, scale4, shift4);

  // 6) BN4+ReLU + transpose-store
  output_kernel<<<32768, 256, 0, stream>>>(Y4, scale4, shift4, out);
}
